// CRW_29867202576498
// MI455X (gfx1250) — matrix operand order repaired and checked by reading, not compiled, not run on hardware
//
#include <hip/hip_runtime.h>
#include <math.h>

// ---------------------------------------------------------------------------
// Problem constants (from reference): B=16, C=128, T=8, N=1024
// ---------------------------------------------------------------------------
#define BB 16
#define CC 128
#define TT 8
#define NN 1024
#define TM1 7                    // T-1 transition matrices
#define INV_TAU 14.2857142857f   // 1/0.07
#define EPSV 1e-20f

typedef __attribute__((ext_vector_type(16))) __bf16 v16bf;
typedef __attribute__((ext_vector_type(8)))  __bf16 v8bf;
typedef __attribute__((ext_vector_type(8)))  float  v8f;

__device__ __forceinline__ v16bf ldfrag(const __bf16* p0, const __bf16* p1) {
    v8bf a = *(const v8bf*)p0;
    v8bf b = *(const v8bf*)p1;
    return __builtin_shufflevector(a, b, 0,1,2,3,4,5,6,7,8,9,10,11,12,13,14,15);
}

__device__ __forceinline__ v8f wmma_bf16(v16bf a, v16bf b, v8f c) {
    return __builtin_amdgcn_wmma_f32_16x16x32_bf16(false, a, false, b, (short)0, c, false, false);
}

__device__ __forceinline__ v8f zero_v8f() {
    v8f z = {0.f,0.f,0.f,0.f,0.f,0.f,0.f,0.f};
    return z;
}

// ---------------------------------------------------------------------------
// K1: L2-normalize feature vectors along C, store transposed bf16:
//     fT[b][t][n][c] = feats[b][c][t][n] / (||feats[b,:,t,n]|| + 1e-12)
// grid (N/256, T, B), block 256.
// ---------------------------------------------------------------------------
__global__ __launch_bounds__(256) void normalize_kernel(const float* __restrict__ feats,
                                                        __bf16* __restrict__ fT) {
    const int n = blockIdx.x * 256 + threadIdx.x;
    const int t = blockIdx.y;
    const int b = blockIdx.z;
    const size_t cstride = (size_t)TT * NN;
    const float* fp = feats + (size_t)b * CC * cstride + (size_t)t * NN + n;
    float ss = 0.f;
    #pragma unroll 8
    for (int c = 0; c < CC; ++c) {
        float v = fp[(size_t)c * cstride];
        ss += v * v;
    }
    const float inv = 1.f / (sqrtf(ss) + 1e-12f);
    __bf16* out = fT + (((size_t)b * TT + t) * NN + n) * CC;
    #pragma unroll 8
    for (int c = 0; c < CC; ++c)
        out[c] = (__bf16)(fp[(size_t)c * cstride] * inv);
}

// ---------------------------------------------------------------------------
// K2: pass A — E = exp((fA . fB^T)/tau) with WMMA bf16, K=128.
//     Writes E ([n][m]) to rightRaw and E^T to leftRaw (vectorized 16B lane
//     stores). Accumulates rowsum(E) -> rsum, colsum(E) -> csum.
// grid (8, 8, B*TM1), block 256 (8 waves; wave tile 32x64).
// ---------------------------------------------------------------------------
__global__ __launch_bounds__(256) void passA_kernel(const __bf16* __restrict__ fT,
                                                    __bf16* __restrict__ rightRaw,
                                                    __bf16* __restrict__ leftRaw,
                                                    float* __restrict__ rsum,
                                                    float* __restrict__ csum) {
    __shared__ __bf16 As[128][40];
    __shared__ __bf16 Bs[128][40];
    __shared__ float rs_l[128];
    __shared__ float cs_l[128];

    const int tid  = threadIdx.x;
    const int lane = tid & 31;
    const int w    = tid >> 5;
    const int wr   = w >> 1;      // 0..3 : 32-row strip
    const int wc   = w & 1;       // 0..1 : 64-col strip
    const int bt = blockIdx.z;
    const int b  = bt / TM1;
    const int t  = bt % TM1;
    const int n0 = blockIdx.y * 128;
    const int m0 = blockIdx.x * 128;

    const __bf16* fA = fT + ((size_t)b * TT + t)     * NN * CC;  // rows n
    const __bf16* fB = fT + ((size_t)b * TT + t + 1) * NN * CC;  // rows m
    __bf16* rightp = rightRaw + (size_t)bt * NN * NN;
    __bf16* leftp  = leftRaw  + (size_t)bt * NN * NN;

    if (tid < 128) { rs_l[tid] = 0.f; cs_l[tid] = 0.f; }

    v8f acc[2][4];
    #pragma unroll
    for (int i = 0; i < 2; ++i)
        #pragma unroll
        for (int j = 0; j < 4; ++j) acc[i][j] = zero_v8f();

    for (int k0 = 0; k0 < CC; k0 += 32) {
        __syncthreads();
        #pragma unroll
        for (int it = 0; it < 2; ++it) {
            int chunk = tid + it * 256;          // 512 chunks of 8 bf16
            int row = chunk >> 2;
            int cc  = (chunk & 3) * 8;
            *(v8bf*)&As[row][cc] = *(const v8bf*)&fA[(size_t)(n0 + row) * CC + k0 + cc];
            *(v8bf*)&Bs[row][cc] = *(const v8bf*)&fB[(size_t)(m0 + row) * CC + k0 + cc];
        }
        __syncthreads();

        v16bf afrag[2], bfrag[4];
        #pragma unroll
        for (int rs = 0; rs < 2; ++rs) {
            int row = wr * 32 + rs * 16 + (lane & 15);
            int kb  = (lane >> 4) * 8;
            afrag[rs] = ldfrag(&As[row][kb], &As[row][kb + 16]);
        }
        #pragma unroll
        for (int ct = 0; ct < 4; ++ct) {
            int m  = wc * 64 + ct * 16 + (lane & 15);
            int kb = (lane >> 4) * 8;
            bfrag[ct] = ldfrag(&Bs[m][kb], &Bs[m][kb + 16]);
        }
        #pragma unroll
        for (int rs = 0; rs < 2; ++rs)
            #pragma unroll
            for (int ct = 0; ct < 4; ++ct)
                acc[rs][ct] = wmma_bf16(afrag[rs], bfrag[ct], acc[rs][ct]);
    }

    // epilogue: exp, stores, LDS sums
    #pragma unroll
    for (int rs = 0; rs < 2; ++rs) {
        #pragma unroll
        for (int ct = 0; ct < 4; ++ct) {
            const int ml = wc * 64 + ct * 16 + (lane & 15);       // local col
            const int nl = wr * 32 + rs * 16 + ((lane >> 4) * 8); // local row base
            v8bf lv;
            #pragma unroll
            for (int r = 0; r < 8; ++r) {
                float e = __expf(acc[rs][ct][r] * INV_TAU);
                rightp[(size_t)(n0 + nl + r) * NN + (m0 + ml)] = (__bf16)e;
                lv[r] = (__bf16)e;
                atomicAdd(&rs_l[nl + r], e);
                atomicAdd(&cs_l[ml], e);
            }
            *(v8bf*)&leftp[(size_t)(m0 + ml) * NN + (n0 + nl)] = lv;  // E^T
        }
    }
    __syncthreads();
    if (tid < 128) {
        atomicAdd(&rsum[(size_t)bt * NN + n0 + tid], rs_l[tid]);
        atomicAdd(&csum[(size_t)bt * NN + m0 + tid], cs_l[tid]);
    }
}

// ---------------------------------------------------------------------------
// K3a: in-place row normalization: buf[row][m] /= sums[row]
// ---------------------------------------------------------------------------
__global__ __launch_bounds__(256) void rownorm_kernel(__bf16* __restrict__ buf,
                                                      const float* __restrict__ sums) {
    size_t base = ((size_t)blockIdx.x * 256 + threadIdx.x) * 8;
    const float inv = 1.f / sums[base >> 10];   // row length == 1024
    v8bf v = *(v8bf*)&buf[base];
    #pragma unroll
    for (int j = 0; j < 8; ++j) v[j] = (__bf16)((float)v[j] * inv);
    *(v8bf*)&buf[base] = v;
}

// ---------------------------------------------------------------------------
// K3b: column-scale into new buffer: dst[..][n] = src[..][n] / scales[slab][n]
//      Used for rightT = leftRaw / rsum  (R^T, all b,t slabs).
// ---------------------------------------------------------------------------
__global__ __launch_bounds__(256) void colscale_kernel(__bf16* __restrict__ dst,
                                                       const __bf16* __restrict__ src,
                                                       const float* __restrict__ scales) {
    size_t base = ((size_t)blockIdx.x * 256 + threadIdx.x) * 8;
    const float* sp = scales + ((base >> 20) << 10) + (base & 1023);
    v8bf v = *(const v8bf*)&src[base];
    #pragma unroll
    for (int j = 0; j < 8; ++j) v[j] = (__bf16)((float)v[j] / sp[j]);
    *(v8bf*)&dst[base] = v;
}

// ---------------------------------------------------------------------------
// K3c: leftT0[b][m][k] = rightRaw[b][t=0][m][k] / csum[b][t=0][k]  (= L0^T)
// ---------------------------------------------------------------------------
__global__ __launch_bounds__(256) void leftT0_kernel(__bf16* __restrict__ dst,
                                                     const __bf16* __restrict__ rightRaw,
                                                     const float* __restrict__ csum) {
    size_t base = ((size_t)blockIdx.x * 256 + threadIdx.x) * 8;   // within B*N*N
    size_t b = base >> 20;
    const size_t nNN = (size_t)NN * NN;
    const __bf16* src = rightRaw + b * TM1 * nNN + (base & (nNN - 1));
    const float*  sp  = csum + b * TM1 * NN + (base & 1023);
    v8bf v = *(const v8bf*)src;
    #pragma unroll
    for (int j = 0; j < 8; ++j) v[j] = (__bf16)((float)v[j] / sp[j]);
    *(v8bf*)&dst[base] = v;
}

// ---------------------------------------------------------------------------
// K4: batched bf16 GEMM  C[b] = A[b] @ BT[b]^T  (1024^3, both operands
//     row-major with contiguous K; staging is pure vectorized b128 copies).
//     Workgroup tile 256x128, 8 waves with 64x64 wave tiles (acc[4][4]):
//     16 ds_load_b128 per 16 WMMAs. K-step 64.
//     Optional outputs: C (row-major) and/or CT (transposed, 16B lane stores).
// grid (8, 4, B), block 256.
// ---------------------------------------------------------------------------
__global__ __launch_bounds__(256) void gemm_bf16_nt(const __bf16* __restrict__ A, size_t strideA,
                                                    const __bf16* __restrict__ BT, size_t strideBT,
                                                    __bf16* __restrict__ C, size_t strideC,
                                                    __bf16* __restrict__ CT, size_t strideCT) {
    __shared__ __bf16 As[256][72];
    __shared__ __bf16 Bs[128][72];

    const int tid  = threadIdx.x;
    const int lane = tid & 31;
    const int w    = tid >> 5;
    const int wr   = w >> 1;      // 0..3 : 64-row strip
    const int wc   = w & 1;       // 0..1 : 64-col strip
    const int n0 = blockIdx.y * 256;
    const int m0 = blockIdx.x * 128;
    A  += (size_t)blockIdx.z * strideA;
    BT += (size_t)blockIdx.z * strideBT;

    v8f acc[4][4];
    #pragma unroll
    for (int i = 0; i < 4; ++i)
        #pragma unroll
        for (int j = 0; j < 4; ++j) acc[i][j] = zero_v8f();

    for (int k0 = 0; k0 < NN; k0 += 64) {
        __syncthreads();
        #pragma unroll
        for (int it = 0; it < 8; ++it) {          // A: 256 rows x 64 k
            int chunk = tid + it * 256;           // 2048 chunks of 8 bf16
            int row = chunk >> 3;                 // 0..255
            int cc  = (chunk & 7) * 8;            // 0..56
            *(v8bf*)&As[row][cc] = *(const v8bf*)&A[(size_t)(n0 + row) * NN + k0 + cc];
        }
        #pragma unroll
        for (int it = 0; it < 4; ++it) {          // BT: 128 rows x 64 k
            int chunk = tid + it * 256;           // 1024 chunks
            int row = chunk >> 3;                 // 0..127
            int cc  = (chunk & 7) * 8;
            *(v8bf*)&Bs[row][cc] = *(const v8bf*)&BT[(size_t)(m0 + row) * NN + k0 + cc];
        }
        __syncthreads();

        if (k0 + 64 < NN) {   // prefetch next K tile (global_prefetch_b8)
            __builtin_prefetch(&A [(size_t)(n0 + tid) * NN + k0 + 64], 0, 1);
            __builtin_prefetch(&BT[(size_t)(m0 + (tid & 127)) * NN + k0 + 64], 0, 1);
        }

        #pragma unroll
        for (int kk = 0; kk < 64; kk += 32) {
            v16bf afrag[4], bfrag[4];
            #pragma unroll
            for (int rs = 0; rs < 4; ++rs) {
                int row = wr * 64 + rs * 16 + (lane & 15);
                int kb  = kk + (lane >> 4) * 8;
                afrag[rs] = ldfrag(&As[row][kb], &As[row][kb + 16]);
            }
            #pragma unroll
            for (int ct = 0; ct < 4; ++ct) {
                int m  = wc * 64 + ct * 16 + (lane & 15);
                int kb = kk + (lane >> 4) * 8;
                bfrag[ct] = ldfrag(&Bs[m][kb], &Bs[m][kb + 16]);
            }
            #pragma unroll
            for (int rs = 0; rs < 4; ++rs)
                #pragma unroll
                for (int ct = 0; ct < 4; ++ct)
                    acc[rs][ct] = wmma_bf16(afrag[rs], bfrag[ct], acc[rs][ct]);
        }
    }

    #pragma unroll
    for (int rs = 0; rs < 4; ++rs) {
        #pragma unroll
        for (int ct = 0; ct < 4; ++ct) {
            const int ml = wc * 64 + ct * 16 + (lane & 15);
            const int nl = wr * 64 + rs * 16 + ((lane >> 4) * 8);
            if (C != nullptr) {
                __bf16* Cp = C + (size_t)blockIdx.z * strideC;
                #pragma unroll
                for (int r = 0; r < 8; ++r)
                    Cp[(size_t)(n0 + nl + r) * NN + (m0 + ml)] = (__bf16)acc[rs][ct][r];
            }
            if (CT != nullptr) {
                __bf16* Tp = CT + (size_t)blockIdx.z * strideCT;
                v8bf tv;
                #pragma unroll
                for (int r = 0; r < 8; ++r) tv[r] = (__bf16)acc[rs][ct][r];
                *(v8bf*)&Tp[(size_t)(m0 + ml) * NN + (n0 + nl)] = tv;
            }
        }
    }
}

// ---------------------------------------------------------------------------
// K5: srow[b][k] = sum_m S[b][k][m] = sum_m ST[b][m][k]  (coalesced over k)
// grid (4, B), block 256.
// ---------------------------------------------------------------------------
__global__ __launch_bounds__(256) void srowcol_kernel(const __bf16* __restrict__ ST,
                                                      float* __restrict__ srow) {
    const int k = blockIdx.x * 256 + threadIdx.x;
    const int b = blockIdx.y;
    const __bf16* p = ST + (size_t)b * NN * NN + k;
    float s = 0.f;
    #pragma unroll 4
    for (int m = 0; m < NN; ++m) s += (float)p[(size_t)m * NN];
    srow[(size_t)b * NN + k] = s;
}

// ---------------------------------------------------------------------------
// K6: per (b,n): diag = Q[n,:].ST[n,:]; rsP = Q[n,:].srow  (all coalesced)
//     lossAcc += -( log(diag+eps) - log(rsP + N*eps) )
// grid (N, B), block 256.
// ---------------------------------------------------------------------------
__global__ __launch_bounds__(256) void loss_kernel(const __bf16* __restrict__ Q,
                                                   const __bf16* __restrict__ ST,
                                                   const float* __restrict__ srow,
                                                   float* __restrict__ lossAcc) {
    __shared__ float red0[256];
    __shared__ float red1[256];
    const int n = blockIdx.x;
    const int b = blockIdx.y;
    const int tid = threadIdx.x;
    const __bf16* Qp = Q  + (size_t)b * NN * NN + (size_t)n * NN;
    const __bf16* Tp = ST + (size_t)b * NN * NN + (size_t)n * NN;
    const float*  rp = srow + (size_t)b * NN;
    float d = 0.f, rs = 0.f;
    for (int k = tid; k < NN; k += 256) {
        float q = (float)Qp[k];
        d  += q * (float)Tp[k];
        rs += q * rp[k];
    }
    red0[tid] = d; red1[tid] = rs;
    __syncthreads();
    for (int o = 128; o > 0; o >>= 1) {
        if (tid < o) { red0[tid] += red0[tid + o]; red1[tid] += red1[tid + o]; }
        __syncthreads();
    }
    if (tid == 0) {
        float term = logf(red0[0] + EPSV) - logf(red1[0] + (float)NN * EPSV);
        atomicAdd(lossAcc, -term);
    }
}

// K7: final scalar
__global__ void final_kernel(const float* __restrict__ lossAcc, float* __restrict__ out) {
    float s = 0.f;
    for (int i = 0; i < 6; ++i) s += lossAcc[i];
    out[0] = s / (6.f * BB * NN);
}

// ---------------------------------------------------------------------------
// Host orchestration
// ---------------------------------------------------------------------------
extern "C" void kernel_launch(void* const* d_in, const int* in_sizes, int n_in,
                              void* d_out, int out_size, void* d_ws, size_t ws_size,
                              hipStream_t stream) {
    (void)in_sizes; (void)n_in; (void)out_size; (void)ws_size;
    const float* feats = (const float*)d_in[0];
    float* out = (float*)d_out;

    char* ws = (char*)d_ws;
    size_t off = 0;
    auto alloc = [&](size_t bytes) { size_t r = off; off = (off + bytes + 255) & ~(size_t)255; return r; };

    const size_t nNN = (size_t)NN * NN;
    __bf16* fT      = (__bf16*)(ws + alloc((size_t)BB * TT * NN * CC * 2));
    __bf16* right   = (__bf16*)(ws + alloc((size_t)BB * TM1 * nNN * 2));   // rightRaw -> right
    __bf16* left    = (__bf16*)(ws + alloc((size_t)BB * TM1 * nNN * 2));   // leftRaw  -> left
    __bf16* rightT  = (__bf16*)(ws + alloc((size_t)BB * TM1 * nNN * 2));   // R^T (B operand of Q-chain)
    __bf16* leftT0  = (__bf16*)(ws + alloc((size_t)BB * nNN * 2));         // L0^T (S0^T)
    float*  rsum    = (float*) (ws + alloc((size_t)BB * TM1 * NN * 4));
    float*  csum    = (float*) (ws + alloc((size_t)BB * TM1 * NN * 4));
    __bf16* Qbuf[2];
    __bf16* STbuf[2];
    Qbuf[0]  = (__bf16*)(ws + alloc((size_t)BB * nNN * 2));
    Qbuf[1]  = (__bf16*)(ws + alloc((size_t)BB * nNN * 2));
    STbuf[0] = (__bf16*)(ws + alloc((size_t)BB * nNN * 2));
    STbuf[1] = (__bf16*)(ws + alloc((size_t)BB * nNN * 2));
    float*  srow    = (float*) (ws + alloc((size_t)BB * NN * 4));
    float*  lossAcc = (float*) (ws + alloc(8 * 4));

    hipMemsetAsync(rsum,    0, (size_t)BB * TM1 * NN * 4, stream);
    hipMemsetAsync(csum,    0, (size_t)BB * TM1 * NN * 4, stream);
    hipMemsetAsync(lossAcc, 0, 8 * 4, stream);

    normalize_kernel<<<dim3(NN / 256, TT, BB), 256, 0, stream>>>(feats, fT);

    // right/left hold RAW E / E^T after this
    passA_kernel<<<dim3(8, 8, BB * TM1), 256, 0, stream>>>(fT, right, left, rsum, csum);

    const int rnBlocks = (int)((size_t)BB * TM1 * nNN / 8 / 256);
    const int t0Blocks = (int)((size_t)BB * nNN / 8 / 256);
    // transposed normalized copies FIRST (they read the raw buffers)
    colscale_kernel<<<dim3(rnBlocks), 256, 0, stream>>>(rightT, left, rsum);   // R^T = E^T / rsum[n]
    leftT0_kernel  <<<dim3(t0Blocks), 256, 0, stream>>>(leftT0, right, csum);  // L0^T = E0 / csum[k]
    // then in-place normalization raw -> softmaxed
    rownorm_kernel<<<dim3(rnBlocks), 256, 0, stream>>>(right, rsum);
    rownorm_kernel<<<dim3(rnBlocks), 256, 0, stream>>>(left,  csum);

    // chain: Q_i = Q_{i-1} @ R_i ; S_i^T = (L_i @ S_{i-1})^T ; loss(Q_i, S_i^T)
    for (int i = 1; i <= 6; ++i) {
        const __bf16* Qsrc  = (i == 1) ? right : Qbuf[(i - 1) & 1];
        size_t sQ           = (i == 1) ? (size_t)TM1 * nNN : nNN;
        const __bf16* STsrc = (i == 1) ? leftT0 : STbuf[(i - 1) & 1];
        __bf16* Qdst  = Qbuf[i & 1];
        __bf16* STdst = STbuf[i & 1];

        gemm_bf16_nt<<<dim3(8, 4, BB), 256, 0, stream>>>(
            Qsrc, sQ, rightT + (size_t)i * nNN, (size_t)TM1 * nNN,
            Qdst, nNN, nullptr, 0);
        gemm_bf16_nt<<<dim3(8, 4, BB), 256, 0, stream>>>(
            left + (size_t)i * nNN, (size_t)TM1 * nNN, STsrc, nNN,
            nullptr, 0, STdst, nNN);
        srowcol_kernel<<<dim3(4, BB), 256, 0, stream>>>(STdst, srow);
        loss_kernel<<<dim3(NN, BB), 256, 0, stream>>>(Qdst, STdst, srow, lossAcc + (i - 1));
    }

    final_kernel<<<1, 1, 0, stream>>>(lossAcc, out);
}
